// SGNS_76828374991214
// MI455X (gfx1250) — matrix operand order repaired and checked by reading, not compiled, not run on hardware
//
#include <hip/hip_runtime.h>
#include <hip/hip_bf16.h>

typedef __attribute__((ext_vector_type(16))) __bf16 v16bf;
typedef __attribute__((ext_vector_type(8)))  float  v8f;
typedef __attribute__((ext_vector_type(4)))  float  v4f;

#define B_TOT 262144
#define KNEG  10
#define DIM   128
#define WAVES_PER_BLOCK 8
#define GROUPS (B_TOT / 16)                 // 16384 groups of 16 rows
#define BLOCKS (GROUPS / WAVES_PER_BLOCK)   // 2048 blocks

__device__ __forceinline__ float logsig(float x) {
    // log(sigmoid(x)) = min(x,0) - log(1 + exp(-|x|))   (numerically stable)
    float ax = __builtin_fabsf(x);
    float m  = fminf(x, 0.0f);
    return m - __logf(1.0f + __expf(-ax));
}

__device__ __forceinline__ v4f ld4(const float* p) { return *(const v4f*)p; }

__global__ void __launch_bounds__(256)
sgns_main(const int* __restrict__ centers,
          const int* __restrict__ pos,
          const int* __restrict__ neg,
          const float* __restrict__ W_in,
          const float* __restrict__ W_out,
          float* __restrict__ partials)
{
    __shared__ float wsum[WAVES_PER_BLOCK];
    const int tid   = threadIdx.x;
    const int wave  = tid >> 5;
    const int lane  = tid & 31;
    const int group = blockIdx.x * WAVES_PER_BLOCK + wave;
    const int gbase = group * 16;
    const int n     = lane & 15;      // tile column == row-within-group
    const int gb    = gbase + n;

    // ---- index gathers (lane l and l+16 duplicate; L2 absorbs) ----
    const int cidx = centers[gb];
    const int pidx = pos[gb];
    int nidx[KNEG];
    const int* nrow = neg + gb * KNEG;
    #pragma unroll
    for (int j = 0; j < KNEG; ++j) nidx[j] = nrow[j];

    // ---- A fragments: v_c rows, 16x32 bf16 per chunk, 4 chunks (D=128) ----
    // Layout (ISA 7.12.2): lanes 0-15 hold K {0..7,16..23}, lanes 16-31 hold {8..15,24..31}
    const float* arow = W_in + (size_t)cidx * DIM;
    const int alo = (lane < 16) ? 0 : 8;
    v16bf afrag[4];
    #pragma unroll
    for (int c = 0; c < 4; ++c) {
        const int base = 32 * c + alo;
        v4f f0 = ld4(arow + base);        // K = alo .. alo+3
        v4f f1 = ld4(arow + base + 4);    // K = alo+4 .. alo+7
        v4f f2 = ld4(arow + base + 16);   // K = 16+alo ..
        v4f f3 = ld4(arow + base + 20);
        v16bf a;
        #pragma unroll
        for (int e = 0; e < 4; ++e) {
            a[e]      = (__bf16)f0[e];
            a[4 + e]  = (__bf16)f1[e];
            a[8 + e]  = (__bf16)f2[e];
            a[12 + e] = (__bf16)f3[e];
        }
        afrag[c] = a;
    }

    // ---- 11 targets: B tile column n = u-vector of row b_n; diag(C) = scores ----
    const int blo = (lane < 16) ? 0 : 8;    // lanes 0-15: K {0..7,16..23}; lanes 16-31: K {8..15,24..31}
    float lane_loss = 0.0f;

    #pragma unroll
    for (int j = 0; j <= KNEG; ++j) {
        const int ridx = (j == 0) ? pidx : nidx[j - 1];
        const float* brow = W_out + (size_t)ridx * DIM;
        v8f acc = {0.f, 0.f, 0.f, 0.f, 0.f, 0.f, 0.f, 0.f};
        #pragma unroll
        for (int c = 0; c < 4; ++c) {
            const int base = 32 * c + blo;
            v4f f0 = ld4(brow + base);
            v4f f1 = ld4(brow + base + 4);
            v4f f2 = ld4(brow + base + 16);
            v4f f3 = ld4(brow + base + 20);
            v16bf b;
            #pragma unroll
            for (int e = 0; e < 4; ++e) {
                b[e]      = (__bf16)f0[e];
                b[4 + e]  = (__bf16)f1[e];
                b[8 + e]  = (__bf16)f2[e];
                b[12 + e] = (__bf16)f3[e];
            }
            acc = __builtin_amdgcn_wmma_f32_16x16x32_bf16(
                /*neg_a=*/false, afrag[c], /*neg_b=*/false, b,
                /*c_mod=*/(short)0, acc, /*reuse_a=*/false, /*reuse_b=*/false);
        }
        // Diagonal extraction: d<8 -> VGPR d lane d; d>=8 -> VGPR d-8 lane d+16
        const int r = (lane < 8) ? lane : (lane - 24);
        float s = acc[0];
        #pragma unroll
        for (int e = 1; e < 8; ++e) s = (r == e) ? acc[e] : s;
        const bool  valid   = (lane < 8) || (lane >= 24);
        const float contrib = (j == 0) ? -logsig(s) : -logsig(-s);
        lane_loss += valid ? contrib : 0.0f;
    }

    // ---- wave32 cross-lane sum (fixed order -> deterministic) ----
    #pragma unroll
    for (int off = 16; off > 0; off >>= 1)
        lane_loss += __shfl_xor(lane_loss, off, 32);

    if (lane == 0) wsum[wave] = lane_loss;
    __syncthreads();
    if (tid == 0) {
        float s = 0.0f;
        #pragma unroll
        for (int w = 0; w < WAVES_PER_BLOCK; ++w) s += wsum[w];
        partials[blockIdx.x] = s;
    }
}

__global__ void __launch_bounds__(256)
sgns_reduce(const float* __restrict__ partials, float* __restrict__ out)
{
    __shared__ float sm[256];
    const int t = threadIdx.x;
    float s = 0.0f;
    #pragma unroll
    for (int i = 0; i < BLOCKS / 256; ++i)   // fixed-order strided sum
        s += partials[t + i * 256];
    sm[t] = s;
    __syncthreads();
    for (int stride = 128; stride > 0; stride >>= 1) {
        if (t < stride) sm[t] += sm[t + stride];
        __syncthreads();
    }
    if (t == 0) out[0] = sm[0] * (1.0f / (float)B_TOT);
}

extern "C" void kernel_launch(void* const* d_in, const int* in_sizes, int n_in,
                              void* d_out, int out_size, void* d_ws, size_t ws_size,
                              hipStream_t stream)
{
    const int*   centers = (const int*)d_in[0];
    const int*   pos     = (const int*)d_in[1];
    const int*   neg     = (const int*)d_in[2];
    const float* W_in    = (const float*)d_in[3];
    const float* W_out   = (const float*)d_in[4];
    float* partials = (float*)d_ws;   // 2048 floats = 8 KB scratch

    sgns_main<<<BLOCKS, 256, 0, stream>>>(centers, pos, neg, W_in, W_out, partials);
    sgns_reduce<<<1, 256, 0, stream>>>(partials, (float*)d_out);

    (void)in_sizes; (void)n_in; (void)out_size; (void)ws_size;
}
